// GNNBlock_76974403879507
// MI455X (gfx1250) — matrix operand order repaired and checked by reading, not compiled, not run on hardware
//
#include <hip/hip_runtime.h>
#include <cstdint>
#include <cstddef>

typedef __attribute__((ext_vector_type(16))) __bf16        v16bf;
typedef __attribute__((ext_vector_type(8)))  float         v8f;
typedef __attribute__((ext_vector_type(8)))  unsigned int  v8u;

#define DFEAT 128
#define WSLOT 16384   // ushorts per packed 128x128 bf16 matrix

// ---------------- scalar helpers ----------------
// RNE conversion (cold prep paths only)
__device__ __forceinline__ unsigned short f2bf_rne(float f) {
  unsigned u = __float_as_uint(f);
  u += 0x7fffu + ((u >> 16) & 1u);
  return (unsigned short)(u >> 16);
}
__device__ __forceinline__ unsigned pack2_rne(float a, float b) {
  return (unsigned)f2bf_rne(a) | ((unsigned)f2bf_rne(b) << 16);
}
// Hot path: truncating pack of two fp32 -> two bf16 in ONE v_perm_b32.
// result = { b[31:16], a[31:16] }  (perm: bytes 0-3 = src1, bytes 4-7 = src0)
__device__ __forceinline__ unsigned pack2(float a, float b) {
  return __builtin_amdgcn_perm(__float_as_uint(b), __float_as_uint(a), 0x07060302u);
}

__device__ __forceinline__ float fast_rcp(float x) {
#if __has_builtin(__builtin_amdgcn_rcpf)
  return __builtin_amdgcn_rcpf(x);      // v_rcp_f32 (TRANS32)
#else
  return 1.0f / x;
#endif
}
__device__ __forceinline__ float fast_exp2(float x) {
#if __has_builtin(__builtin_amdgcn_exp2f)
  return __builtin_amdgcn_exp2f(x);     // v_exp_f32 (TRANS32)
#else
  return exp2f(x);
#endif
}

// gelu(x) = 0.5*x*(1+erf(x/sqrt2)); erf via Abramowitz-Stegun 7.1.26
// (|err| <= 1.5e-7; well below bf16 GEMM noise). ~14 VALU, 2 TRANS, no branch.
__device__ __forceinline__ float gelu_f(float x) {
  const float z  = x * 0.7071067811865476f;
  const float az = __builtin_fabsf(z);
  const float t  = fast_rcp(__builtin_fmaf(0.3275911f, az, 1.0f));
  float p = __builtin_fmaf(1.061405429f, t, -1.453152027f);
  p = __builtin_fmaf(p, t, 1.421413741f);
  p = __builtin_fmaf(p, t, -0.284496736f);
  p = __builtin_fmaf(p, t, 0.254829592f);
  p *= t;
  const float e = fast_exp2(az * az * -1.4426950408889634f);  // e^{-z^2}
  const float erf_abs = __builtin_fmaf(-p, e, 1.0f);
  const float erf_z   = __builtin_copysignf(erf_abs, z);
  return 0.5f * x * (1.0f + erf_z);
}

// ---------------- WMMA fragment loaders ----------------
// A fragment 16x32 bf16 from an fp32 row (edge_feats, m): 8 v_perm total.
__device__ __forceinline__ v16bf load_A_f32(const float* __restrict__ row,
                                            int kc, int half) {
  const float* p0 = row + kc * 32 + half * 8;
  const float* p1 = p0 + 16;
  float4 a = *(const float4*)(p0);
  float4 b = *(const float4*)(p0 + 4);
  float4 c = *(const float4*)(p1);
  float4 d = *(const float4*)(p1 + 4);
  v8u u;
  u[0] = pack2(a.x, a.y); u[1] = pack2(a.z, a.w);
  u[2] = pack2(b.x, b.y); u[3] = pack2(b.z, b.w);
  u[4] = pack2(c.x, c.y); u[5] = pack2(c.z, c.w);
  u[6] = pack2(d.x, d.y); u[7] = pack2(d.z, d.w);
  return __builtin_bit_cast(v16bf, u);
}

// A fragment from a bf16 row (pre-converted nodes in global, or LDS scratch):
// two 16B vector loads, zero conversion ALU. Addrspace inferred per call site.
__device__ __forceinline__ v16bf load_A_bf16(const unsigned short* row,
                                             int kc, int half) {
  const unsigned short* p0 = row + kc * 32 + half * 8;
  const uint4 q0 = *(const uint4*)(p0);
  const uint4 q1 = *(const uint4*)(p0 + 16);
  v8u u;
  u[0] = q0.x; u[1] = q0.y; u[2] = q0.z; u[3] = q0.w;
  u[4] = q1.x; u[5] = q1.y; u[6] = q1.z; u[7] = q1.w;
  return __builtin_bit_cast(v16bf, u);
}

// B fragment 32x16 bf16 from pre-swizzled LDS weights.
__device__ __forceinline__ v16bf load_B_lds(const unsigned short* wbase,
                                            int nt, int kc, int lane) {
  const unsigned short* p = wbase + ((nt * 4 + kc) << 9) + (lane << 4);
  const uint4 q0 = *(const uint4*)(p);
  const uint4 q1 = *(const uint4*)(p + 8);
  v8u u;
  u[0] = q0.x; u[1] = q0.y; u[2] = q0.z; u[3] = q0.w;
  u[4] = q1.x; u[5] = q1.y; u[6] = q1.z; u[7] = q1.w;
  return __builtin_bit_cast(v16bf, u);
}

__device__ __forceinline__ void load_A4_f32(v16bf A[4],
                                            const float* __restrict__ row,
                                            int half) {
#pragma unroll
  for (int kc = 0; kc < 4; ++kc) A[kc] = load_A_f32(row, kc, half);
}
__device__ __forceinline__ void load_A4_bf16(v16bf A[4],
                                             const unsigned short* row,
                                             int half) {
#pragma unroll
  for (int kc = 0; kc < 4; ++kc) A[kc] = load_A_bf16(row, kc, half);
}

// [16,128] += [16,128(K)] x [128,128] via 8 N-tiles x 4 K-chunks of WMMA bf16
__device__ __forceinline__ void gemm_acc(const v16bf A[4],
                                         const unsigned short* wbase,
                                         int lane, v8f acc[8]) {
#pragma unroll
  for (int nt = 0; nt < 8; ++nt) {
#pragma unroll
    for (int kc = 0; kc < 4; ++kc) {
      v16bf B = load_B_lds(wbase, nt, kc, lane);
      acc[nt] = __builtin_amdgcn_wmma_f32_16x16x32_bf16(
          false, A[kc], false, B, (short)0, acc[nt], false, false);
    }
  }
}

__device__ __forceinline__ void init_acc1(v8f acc[8],
                                          const float* __restrict__ b, int col) {
#pragma unroll
  for (int nt = 0; nt < 8; ++nt) {
    float v = b[nt * 16 + col];
#pragma unroll
    for (int r = 0; r < 8; ++r) acc[nt][r] = v;
  }
}
__device__ __forceinline__ void init_acc2(v8f acc[8],
                                          const float* __restrict__ b0,
                                          const float* __restrict__ b1, int col) {
#pragma unroll
  for (int nt = 0; nt < 8; ++nt) {
    int n = nt * 16 + col;
    float v = b0[n] + b1[n];
#pragma unroll
    for (int r = 0; r < 8; ++r) acc[nt][r] = v;
  }
}
__device__ __forceinline__ void init_acc3(v8f acc[8],
                                          const float* __restrict__ b0,
                                          const float* __restrict__ b1,
                                          const float* __restrict__ b2, int col) {
#pragma unroll
  for (int nt = 0; nt < 8; ++nt) {
    int n = nt * 16 + col;
    float v = b0[n] + b1[n] + b2[n];
#pragma unroll
    for (int r = 0; r < 8; ++r) acc[nt][r] = v;
  }
}

// C layout (VGPR r: M = r + 8*half, N = nt*16 + lane%16) -> LDS bf16 [M][128].
// Truncating bf16 store: store(trunc(srl x,16)) folds to ds_store_b16_d16_hi.
__device__ __forceinline__ void store_C_lds(unsigned short* scr,
                                            const v8f acc[8], int lane,
                                            bool dogelu) {
  const int half = lane >> 4, col = lane & 15;
#pragma unroll
  for (int nt = 0; nt < 8; ++nt) {
    const int K = nt * 16 + col;
#pragma unroll
    for (int r = 0; r < 8; ++r) {
      float v = acc[nt][r];
      if (dogelu) v = gelu_f(v);
      scr[(half * 8 + r) * DFEAT + K] =
          (unsigned short)(__float_as_uint(v) >> 16);
    }
  }
  asm volatile("" ::: "memory");   // keep DS store->load order (DS in-order per wave)
}

// ---------------- kernels ----------------

// Pack 9 fp32 [128,128] weights into bf16 fragments in WMMA-B lane order.
__global__ void prep_pack(const float* __restrict__ W0, const float* __restrict__ W1,
                          const float* __restrict__ W2, const float* __restrict__ W3,
                          const float* __restrict__ W4, const float* __restrict__ W5,
                          const float* __restrict__ W6, const float* __restrict__ W7,
                          const float* __restrict__ W8, unsigned short* __restrict__ pk) {
  int tid = blockIdx.x * blockDim.x + threadIdx.x;
  if (tid >= 9 * WSLOT) return;
  int w    = tid >> 14;
  int rem  = tid & (WSLOT - 1);
  int frag = rem >> 9;
  int lane = (rem >> 4) & 31;
  int j    = rem & 15;
  int nt = frag >> 2, kc = frag & 3;
  int K = kc * 32 + (lane >> 4) * 8 + (j & 7) + (j >> 3) * 16;   // table K order per lane
  int n = nt * 16 + (lane & 15);
  const float* W;
  switch (w) {
    case 0: W = W0; break; case 1: W = W1; break; case 2: W = W2; break;
    case 3: W = W3; break; case 4: W = W4; break; case 5: W = W5; break;
    case 6: W = W6; break; case 7: W = W7; break; default: W = W8; break;
  }
  pk[tid] = f2bf_rne(W[K * DFEAT + n]);
}

// Pre-convert node features to bf16 rows (RNE): 8 floats -> uint4 per thread.
__global__ void prep_nodes(const float* __restrict__ nf,
                           unsigned short* __restrict__ out, int n8) {
  int i = blockIdx.x * blockDim.x + threadIdx.x;
  int stride = gridDim.x * blockDim.x;
  for (; i < n8; i += stride) {
    const float4 a = ((const float4*)nf)[2 * i];
    const float4 b = ((const float4*)nf)[2 * i + 1];
    uint4 o;
    o.x = pack2_rne(a.x, a.y); o.y = pack2_rne(a.z, a.w);
    o.z = pack2_rne(b.x, b.y); o.w = pack2_rne(b.z, b.w);
    ((uint4*)out)[i] = o;
  }
}

__global__ void zero_f32(float* __restrict__ p, int n4) {
  int i = blockIdx.x * blockDim.x + threadIdx.x;
  int stride = gridDim.x * blockDim.x;
  float4 z = make_float4(0.f, 0.f, 0.f, 0.f);
  for (; i < n4; i += stride) ((float4*)p)[i] = z;
}

// Fused edge pipeline: per wave, a 16-edge tile runs 6 GEMMs + scatter.
__global__ void __launch_bounds__(256, 1)
edge_kernel(const float* __restrict__ node_feats,
            const unsigned short* __restrict__ node_bf16,
            const float* __restrict__ edge_feats,
            const int* __restrict__ src_idx, const int* __restrict__ dst_idx,
            const unsigned short* __restrict__ Wpk,
            const float* __restrict__ b_ae1, const float* __restrict__ b_ae2,
            const float* __restrict__ b_src, const float* __restrict__ b_dst,
            const float* __restrict__ b_p1,  const float* __restrict__ b_p2,
            float* __restrict__ m, int E, int numTiles) {
  extern __shared__ __align__(16) unsigned short smem[];
  // Stage 6 bf16 weight matrices (W_ae1,W_ae2,W_src,W_dst,W_p1,W_p2) -> 192KB LDS
  {
    const unsigned* g = (const unsigned*)Wpk;
    unsigned* s = (unsigned*)smem;
    for (int i = threadIdx.x; i < 6 * WSLOT / 2; i += blockDim.x) s[i] = g[i];
  }
  __syncthreads();

  const int waveId = threadIdx.x >> 5;
  const int lane   = threadIdx.x & 31;
  const int half   = lane >> 4;
  const int row    = lane & 15;
  const int col    = lane & 15;
  unsigned short* scr = smem + 6 * WSLOT + waveId * (16 * DFEAT);

  const unsigned short* Wae1 = smem + 0 * WSLOT;
  const unsigned short* Wae2 = smem + 1 * WSLOT;
  const unsigned short* Wsrc = smem + 2 * WSLOT;
  const unsigned short* Wdst = smem + 3 * WSLOT;
  const unsigned short* Wp1  = smem + 4 * WSLOT;
  const unsigned short* Wp2  = smem + 5 * WSLOT;

  int tile = blockIdx.x * 8 + waveId;
  const int stride = gridDim.x * 8;

  for (; tile < numTiles; tile += stride) {
    int e = tile * 16 + row;
    if (e >= E) e = E - 1;                      // clamp tail (extra rows discarded)
    __builtin_prefetch(edge_feats + (size_t)(tile + stride) * 16 * DFEAT, 0, 1);

    v16bf A[4];
    v8f acc[8];

    // ---- t1 = gelu(edge_feats @ W_ae1 + b_ae1) ----
    load_A4_f32(A, edge_feats + (size_t)e * DFEAT, half);
    init_acc1(acc, b_ae1, col);
    gemm_acc(A, Wae1, lane, acc);
    store_C_lds(scr, acc, lane, true);

    // ---- s = t1@W_ae2 + h_src@W_src + h_dst@W_dst + biases ----
    load_A4_bf16(A, scr + row * DFEAT, half);
    init_acc3(acc, b_ae2, b_src, b_dst, col);
    gemm_acc(A, Wae2, lane, acc);

    const int si = src_idx[e];
    load_A4_bf16(A, node_bf16 + (size_t)si * DFEAT, half);
    gemm_acc(A, Wsrc, lane, acc);

    const int di = dst_idx[e];
    load_A4_bf16(A, node_bf16 + (size_t)di * DFEAT, half);
    gemm_acc(A, Wdst, lane, acc);

    // ---- phi: gelu(s) -> W_p1+b_p1 -> gelu -> W_p2+b_p2 ----
    store_C_lds(scr, acc, lane, true);          // gelu(s)
    load_A4_bf16(A, scr + row * DFEAT, half);
    init_acc1(acc, b_p1, col);
    gemm_acc(A, Wp1, lane, acc);

    store_C_lds(scr, acc, lane, true);          // gelu(.)
    load_A4_bf16(A, scr + row * DFEAT, half);
    init_acc1(acc, b_p2, col);
    gemm_acc(A, Wp2, lane, acc);                // acc = h_e_emb [16,128], C layout

    // ---- msg = h_src * h_e_emb ; scatter-add into m[dst] ----
    const int ebase = tile * 16 + half * 8;
    int su[8], du[8];
#pragma unroll
    for (int r = 0; r < 8; ++r) {
      int ee = ebase + r; if (ee >= E) ee = E - 1;
      su[r] = src_idx[ee]; du[r] = dst_idx[ee];
    }
#pragma unroll
    for (int nt = 0; nt < 8; ++nt) {
      const int n = nt * 16 + col;
#pragma unroll
      for (int r = 0; r < 8; ++r) {
        if (ebase + r < E) {
          float hs = node_feats[(size_t)su[r] * DFEAT + n];
          atomicAdd(&m[(size_t)du[r] * DFEAT + n], hs * acc[nt][r]);
        }
      }
    }
  }
}

// Node update: out = gelu((h@W_td + b_td + m@W_tm + b_tm) @ W_t + b_t)
__global__ void __launch_bounds__(256, 1)
node_kernel(const unsigned short* __restrict__ node_bf16,
            const float* __restrict__ m,
            const unsigned short* __restrict__ Wpk,
            const float* __restrict__ b_td, const float* __restrict__ b_tm,
            const float* __restrict__ b_t, float* __restrict__ out,
            int N, int numTiles) {
  extern __shared__ __align__(16) unsigned short smem[];
  {
    const unsigned* g = (const unsigned*)(Wpk + 6 * WSLOT);   // W_td,W_tm,W_t
    unsigned* s = (unsigned*)smem;
    for (int i = threadIdx.x; i < 3 * WSLOT / 2; i += blockDim.x) s[i] = g[i];
  }
  __syncthreads();

  const int waveId = threadIdx.x >> 5;
  const int lane   = threadIdx.x & 31;
  const int half   = lane >> 4;
  const int row    = lane & 15;
  const int col    = lane & 15;
  unsigned short* scr = smem + 3 * WSLOT + waveId * (16 * DFEAT);

  const unsigned short* Wtd = smem + 0 * WSLOT;
  const unsigned short* Wtm = smem + 1 * WSLOT;
  const unsigned short* Wt  = smem + 2 * WSLOT;

  int tile = blockIdx.x * 8 + waveId;
  const int stride = gridDim.x * 8;

  for (; tile < numTiles; tile += stride) {
    int nd = tile * 16 + row;
    if (nd >= N) nd = N - 1;

    v16bf A[4];
    v8f acc[8];

    load_A4_bf16(A, node_bf16 + (size_t)nd * DFEAT, half);
    init_acc2(acc, b_td, b_tm, col);
    gemm_acc(A, Wtd, lane, acc);

    load_A4_f32(A, m + (size_t)nd * DFEAT, half);
    gemm_acc(A, Wtm, lane, acc);                // acc = pre (no activation)

    store_C_lds(scr, acc, lane, false);
    load_A4_bf16(A, scr + row * DFEAT, half);
    init_acc1(acc, b_t, col);
    gemm_acc(A, Wt, lane, acc);

    const int nbase = tile * 16 + half * 8;
#pragma unroll
    for (int nt = 0; nt < 8; ++nt) {
      const int n = nt * 16 + col;
#pragma unroll
      for (int r = 0; r < 8; ++r) {
        if (nbase + r < N)
          out[(size_t)(nbase + r) * DFEAT + n] = gelu_f(acc[nt][r]);
      }
    }
  }
}

// ---------------- host launcher ----------------
extern "C" void kernel_launch(void* const* d_in, const int* in_sizes, int n_in,
                              void* d_out, int out_size, void* d_ws, size_t ws_size,
                              hipStream_t stream) {
  const float* node_feats = (const float*)d_in[0];
  const float* edge_feats = (const float*)d_in[1];
  const int*   src_idx    = (const int*)d_in[2];
  const int*   dst_idx    = (const int*)d_in[3];
  const float* W_ae1 = (const float*)d_in[4];  const float* b_ae1 = (const float*)d_in[5];
  const float* W_ae2 = (const float*)d_in[6];  const float* b_ae2 = (const float*)d_in[7];
  const float* W_src = (const float*)d_in[8];  const float* b_src = (const float*)d_in[9];
  const float* W_dst = (const float*)d_in[10]; const float* b_dst = (const float*)d_in[11];
  const float* W_p1  = (const float*)d_in[12]; const float* b_p1  = (const float*)d_in[13];
  const float* W_p2  = (const float*)d_in[14]; const float* b_p2  = (const float*)d_in[15];
  const float* W_td  = (const float*)d_in[16]; const float* b_td  = (const float*)d_in[17];
  const float* W_tm  = (const float*)d_in[18]; const float* b_tm  = (const float*)d_in[19];
  const float* W_t   = (const float*)d_in[20]; const float* b_t   = (const float*)d_in[21];

  const int N = in_sizes[0] / DFEAT;
  const int E = in_sizes[1] / DFEAT;

  // workspace layout
  unsigned short* pk = (unsigned short*)d_ws;                      // 9*16384 bf16
  char* wsp = (char*)d_ws + (size_t)9 * WSLOT * 2;                 // 294912 B
  float* m = (float*)wsp;                                          // [N,128] fp32
  unsigned short* node_bf16 =
      (unsigned short*)(wsp + (size_t)N * DFEAT * sizeof(float));  // [N,128] bf16

  // 1) pack weights to bf16 fragment layout
  prep_pack<<<(9 * WSLOT + 255) / 256, 256, 0, stream>>>(
      W_ae1, W_ae2, W_src, W_dst, W_p1, W_p2, W_td, W_tm, W_t, pk);

  // 2) pre-convert node features to bf16 rows + zero message accumulator
  prep_nodes<<<((N * DFEAT / 8) + 255) / 256, 256, 0, stream>>>(
      node_feats, node_bf16, N * DFEAT / 8);
  zero_f32<<<256, 256, 0, stream>>>(m, N * DFEAT / 4);

  // 3) fused edge pipeline (6 GEMMs + gather/scatter), big-LDS config
  const int edge_lds = 6 * WSLOT * 2 + 8 * 16 * DFEAT * 2;         // 229376 B
  hipFuncSetAttribute((const void*)edge_kernel,
                      hipFuncAttributeMaxDynamicSharedMemorySize, edge_lds);
  const int numTilesE = (E + 15) / 16;
  edge_kernel<<<dim3(1280), dim3(256), edge_lds, stream>>>(
      node_feats, node_bf16, edge_feats, src_idx, dst_idx, pk,
      b_ae1, b_ae2, b_src, b_dst, b_p1, b_p2, m, E, numTilesE);

  // 4) node update (3 GEMMs)
  const int node_lds = 3 * WSLOT * 2 + 8 * 16 * DFEAT * 2;         // 131072 B
  hipFuncSetAttribute((const void*)node_kernel,
                      hipFuncAttributeMaxDynamicSharedMemorySize, node_lds);
  const int numTilesN = (N + 15) / 16;
  node_kernel<<<dim3(400), dim3(256), node_lds, stream>>>(
      node_bf16, m, pk, b_td, b_tm, b_t, (float*)d_out, N, numTilesN);
}
